// SecondOrderLatentSDE_2920577761547
// MI455X (gfx1250) — matrix operand order repaired and checked by reading, not compiled, not run on hardware
//
#include <hip/hip_runtime.h>
#include <hip/hip_bf16.h>

typedef __attribute__((ext_vector_type(16))) _Float16 v16h;
typedef __attribute__((ext_vector_type(8)))  _Float16 v8h;
typedef __attribute__((ext_vector_type(8)))  float    v8f;

#define SA 1032              // padded f16 row stride (1024 + 8) to spread LDS banks
#define ROWS 32              // batch rows per workgroup
#define BDIM 513             // 2L+1
#define BATCH 16384

union AFrag { v16h v; v8h h[2]; };

// ---------------------------------------------------------------------------
// Pack a row-major f32 weight (K x N) into per-WMMA-tile f16 B-fragments.
// Tile (k_tile, n_tile) -> 512 contiguous f16: lane-major; lane holds column
// n_tile*16 + lane%16, slots 0..7 -> K = half*8 + s, slots 8..15 -> K = 16 + half*8 + (s-8).
// ---------------------------------------------------------------------------
__global__ void pack_weight(const float* __restrict__ W, _Float16* __restrict__ Wp,
                            int K, int N) {
    const int total = K * N;
    const int KT = K >> 5;
    for (int e = blockIdx.x * blockDim.x + threadIdx.x; e < total;
         e += gridDim.x * blockDim.x) {
        int tile   = e >> 9;       // / 512
        int within = e & 511;
        int lane   = within >> 4;
        int s      = within & 15;
        int k_tile = tile % KT;
        int n_tile = tile / KT;
        int k = (k_tile << 5) + ((lane >> 4) << 3) + (s & 7) + ((s >> 3) << 4);
        int n = (n_tile << 4) + (lane & 15);
        Wp[e] = (_Float16)W[(size_t)k * N + n];
    }
}

// A fragment (16x32 f16) from LDS: M = lane%16; slots 0..7 -> K = half*8 + s,
// slots 8..15 -> K = 16 + half*8 + (s-8).  Two b128 ds loads per lane.
__device__ __forceinline__ v16h load_a_frag(const _Float16* base, int stride) {
    int lane = threadIdx.x & 31;
    const _Float16* p = base + (lane & 15) * stride + ((lane >> 4) << 3);
    AFrag f;
    f.h[0] = *(const v8h*)p;
    f.h[1] = *(const v8h*)(p + 16);
    return f.v;
}

// B fragment: one 32-byte contiguous load per lane from the packed weights.
__device__ __forceinline__ v16h load_b_frag(const _Float16* tile_base) {
    int lane = threadIdx.x & 31;
    return *(const v16h*)(tile_base + (lane << 4));
}

// hidden = tanh(A[32 x Kdim] @ W[Kdim x 1024] + b) -> f16 LDS (stride SA).
// Each wave owns an n-tile; one B-fragment load feeds BOTH m-tiles (2 WMMAs
// per load, two independent accumulation chains for ILP).
__device__ __forceinline__ void mlp_hidden(const _Float16* __restrict__ As,
                                           const _Float16* __restrict__ Wp,
                                           const float* __restrict__ bias,
                                           int Kdim, _Float16* __restrict__ Hs,
                                           int wave, int lane) {
    const int KT = Kdim >> 5;
    for (int n_tile = wave; n_tile < 64; n_tile += 8) {
        const _Float16* wt = Wp + (size_t)n_tile * KT * 512;
        v8f acc0 = {}, acc1 = {};
        for (int kt = 0; kt < KT; ++kt) {
            v16h b  = load_b_frag(wt + (kt << 9));
            v16h a0 = load_a_frag(As + (kt << 5), SA);
            v16h a1 = load_a_frag(As + 16 * SA + (kt << 5), SA);
            acc0 = __builtin_amdgcn_wmma_f32_16x16x32_f16(false, a0, false, b,
                                                          (short)0, acc0, false, false);
            acc1 = __builtin_amdgcn_wmma_f32_16x16x32_f16(false, a1, false, b,
                                                          (short)0, acc1, false, false);
        }
        const int n  = (n_tile << 4) + (lane & 15);
        const float bv = bias[n];
        const int mb = (lane >> 4) << 3;                  // C/D: M = r + 8*(lane/16)
#pragma unroll
        for (int r = 0; r < 8; ++r) {
            Hs[(mb + r) * SA + n]      = (_Float16)tanhf(acc0[r] + bv);
            Hs[(16 + mb + r) * SA + n] = (_Float16)tanhf(acc1[r] + bv);
        }
    }
}

// out[2] = H[32 x 1024] @ W2[1024 x 256] + b2 for one n-tile, both m-tiles.
__device__ __forceinline__ void gemm2_pair(const _Float16* __restrict__ Hs,
                                           const _Float16* __restrict__ Wp,
                                           const float* __restrict__ bias,
                                           int n_tile, int lane, v8f* out) {
    const _Float16* wt = Wp + (size_t)n_tile * 32 * 512;  // KT = 32
    v8f acc0 = {}, acc1 = {};
    for (int kt = 0; kt < 32; ++kt) {
        v16h b  = load_b_frag(wt + (kt << 9));
        v16h a0 = load_a_frag(Hs + (kt << 5), SA);
        v16h a1 = load_a_frag(Hs + 16 * SA + (kt << 5), SA);
        acc0 = __builtin_amdgcn_wmma_f32_16x16x32_f16(false, a0, false, b,
                                                      (short)0, acc0, false, false);
        acc1 = __builtin_amdgcn_wmma_f32_16x16x32_f16(false, a1, false, b,
                                                      (short)0, acc1, false, false);
    }
    const float bv = bias[(n_tile << 4) + (lane & 15)];
#pragma unroll
    for (int r = 0; r < 8; ++r) { acc0[r] += bv; acc1[r] += bv; }
    out[0] = acc0;
    out[1] = acc1;
}

// ---------------------------------------------------------------------------
// Fused: stage [pm|ctx] -> hiddenH -> prior(regs) -> hiddenC (reuse LDS)
//        -> posterior(regs) -> drift[256:512]=post, drift[512]=dkl
// ---------------------------------------------------------------------------
__global__ __launch_bounds__(256) void sde_main(
    const float* __restrict__ z, const float* __restrict__ ts,
    const float* __restrict__ tscalar, const float* __restrict__ ctx,
    const float* __restrict__ diffusion,
    const float* __restrict__ bh1, const float* __restrict__ bh2,
    const float* __restrict__ bc1, const float* __restrict__ bc2,
    const _Float16* __restrict__ Wh1p, const _Float16* __restrict__ Wc1p,
    const _Float16* __restrict__ Wh2p, const _Float16* __restrict__ Wc2p,
    float* __restrict__ drift) {
    extern __shared__ _Float16 smem[];
    _Float16* As  = smem;                     // 32 x SA  (A operand, both MLPs)
    _Float16* Hs  = smem + ROWS * SA;         // 32 x SA  (hidden, reused)
    float*    dkl = (float*)(smem + 2 * ROWS * SA);

    const int tid  = threadIdx.x;
    const int lane = tid & 31;
    const int wave = tid >> 5;
    const int row0 = blockIdx.x * ROWS;

    if (tid < ROWS) dkl[tid] = 0.0f;

    // i = min(searchsorted(ts, t, 'right'), T-1)  -- count of ts[j] <= t
    const float tv = tscalar[0];
    int ictx = 0;
    for (int j = 0; j < 16; ++j) ictx += (ts[j] <= tv) ? 1 : 0;
    if (ictx > 15) ictx = 15;

    // stage A = [pm | ctx[:, i]] as f16
    for (int e = tid; e < ROWS * 1024; e += 256) {
        const int m = e >> 10, k = e & 1023;
        const size_t row = (size_t)(row0 + m);
        const float v = (k < 512) ? z[row * BDIM + k]
                                  : ctx[(row * 16 + ictx) * 512 + (k - 512)];
        As[m * SA + k] = (_Float16)v;
    }
    __syncthreads();

    // prior hidden + prior output (registers: 2 n-tiles x 2 m-tiles per wave)
    mlp_hidden(As, Wh1p, bh1, 512, Hs, wave, lane);
    __syncthreads();
    v8f accP[4];
#pragma unroll
    for (int j = 0; j < 2; ++j)
        gemm2_pair(Hs, Wh2p, bh2, wave + 8 * j, lane, &accP[2 * j]);
    __syncthreads();

    // posterior hidden reuses Hs
    mlp_hidden(As, Wc1p, bc1, 1024, Hs, wave, lane);
    __syncthreads();

#pragma unroll
    for (int j = 0; j < 2; ++j) {
        const int n_tile = wave + 8 * j;
        v8f post[2];
        gemm2_pair(Hs, Wc2p, bc2, n_tile, lane, post);
        const int n = (n_tile << 4) + (lane & 15);
        const float sig = diffusion[n];
        const float sd  = (fabsf(sig) > 1e-7f) ? sig : copysignf(1e-7f, sig);
        const float inv = 1.0f / sd;
#pragma unroll
        for (int m = 0; m < 2; ++m) {
            const int mb = (m << 4) + ((lane >> 4) << 3);
#pragma unroll
            for (int r = 0; r < 8; ++r) {
                const int rowl = mb + r;
                const float pv = post[m][r];
                drift[(size_t)(row0 + rowl) * BDIM + 256 + n] = pv;
                float d  = (accP[2 * j + m][r] - pv) * inv;
                float sq = d * d;
#pragma unroll
                for (int off = 1; off < 16; off <<= 1) sq += __shfl_xor(sq, off, 32);
                if ((lane & 15) == 0) atomicAdd(&dkl[rowl], sq);
            }
        }
    }
    __syncthreads();
    if (tid < ROWS) drift[(size_t)(row0 + tid) * BDIM + 512] = dkl[tid];
}

// bandwidth-trivial outputs: position drift cols [0,256), all of diffusion_out
__global__ void sde_elem(const float* __restrict__ z,
                         const float* __restrict__ inv_mass,
                         const float* __restrict__ diffusion,
                         float* __restrict__ drift, float* __restrict__ diffout) {
    const size_t idx = (size_t)blockIdx.x * blockDim.x + threadIdx.x;
    const size_t total = (size_t)BATCH * BDIM;
    if (idx >= total) return;
    const int c = (int)(idx % BDIM);
    const size_t row = idx / BDIM;
    diffout[idx] = (c >= 256 && c < 512) ? diffusion[c - 256] : 0.0f;
    if (c < 256) drift[idx] = inv_mass[c] * z[row * BDIM + 256 + c];
}

extern "C" void kernel_launch(void* const* d_in, const int* in_sizes, int n_in,
                              void* d_out, int out_size, void* d_ws, size_t ws_size,
                              hipStream_t stream) {
    const float* z        = (const float*)d_in[0];
    const float* ts       = (const float*)d_in[1];
    const float* t        = (const float*)d_in[2];
    const float* ctx      = (const float*)d_in[3];
    const float* inv_mass = (const float*)d_in[4];
    const float* diffusion= (const float*)d_in[5];
    const float* Wc1      = (const float*)d_in[6];
    const float* bc1      = (const float*)d_in[7];
    const float* Wc2      = (const float*)d_in[8];
    const float* bc2      = (const float*)d_in[9];
    const float* Wh1      = (const float*)d_in[10];
    const float* bh1      = (const float*)d_in[11];
    const float* Wh2      = (const float*)d_in[12];
    const float* bh2      = (const float*)d_in[13];

    float* drift   = (float*)d_out;
    float* diffout = drift + (size_t)BATCH * BDIM;

    // workspace: packed f16 weights (4 MB total)
    _Float16* ws   = (_Float16*)d_ws;
    _Float16* Wh1p = ws;                       // 512*1024
    _Float16* Wc1p = Wh1p + 524288;            // 1024*1024
    _Float16* Wh2p = Wc1p + 1048576;           // 1024*256
    _Float16* Wc2p = Wh2p + 262144;            // 1024*256

    pack_weight<<<1024, 256, 0, stream>>>(Wh1, Wh1p, 512, 1024);
    pack_weight<<<1024, 256, 0, stream>>>(Wc1, Wc1p, 1024, 1024);
    pack_weight<<<1024, 256, 0, stream>>>(Wh2, Wh2p, 1024, 256);
    pack_weight<<<1024, 256, 0, stream>>>(Wc2, Wc2p, 1024, 256);

    const size_t total = (size_t)BATCH * BDIM;
    sde_elem<<<(int)((total + 255) / 256), 256, 0, stream>>>(z, inv_mass, diffusion,
                                                             drift, diffout);

    const size_t shmem = (size_t)2 * ROWS * SA * sizeof(_Float16) + ROWS * sizeof(float);
    sde_main<<<BATCH / ROWS, 256, shmem, stream>>>(z, ts, t, ctx, diffusion,
                                                   bh1, bh2, bc1, bc2,
                                                   Wh1p, Wc1p, Wh2p, Wc2p, drift);
}
